// BagOfWordsMLP_65386582114769
// MI455X (gfx1250) — matrix operand order repaired and checked by reading, not compiled, not run on hardware
//
#include <hip/hip_runtime.h>
#include <hip/hip_bf16.h>

typedef __bf16 bf16_t;
typedef bf16_t v16bf __attribute__((ext_vector_type(16)));
typedef bf16_t v8bf  __attribute__((ext_vector_type(8)));
typedef bf16_t v4bf  __attribute__((ext_vector_type(4)));
typedef float  v8f   __attribute__((ext_vector_type(8)));

#define BATCH 1024
#define SEQ   512
#define H1D   1024
#define H2D   512
#define NCLS  20

// ---------------------------------------------------------------------------
// Stage 1: h1[b,:] = relu( sum_s W1[x[b,s],:] + b1 ), output bf16 for WMMA.
// One block per batch row; tokens staged in LDS; each thread owns 4 hidden
// dims as a float4, so each iteration the block streams one whole 4KB W1 row
// fully coalesced. ~2GB total gathered traffic -> bandwidth bound, mostly L2.
// ---------------------------------------------------------------------------
__global__ __launch_bounds__(256) void k1_bow_fc1(
    const int* __restrict__ x, const float* __restrict__ W1,
    const float* __restrict__ b1, bf16_t* __restrict__ h1) {
  __shared__ int toks[SEQ];
  const int b = blockIdx.x;
  const int t = threadIdx.x;
  toks[t]       = x[b * SEQ + t];
  toks[t + 256] = x[b * SEQ + t + 256];
  __syncthreads();

  float a0 = 0.f, a1 = 0.f, a2 = 0.f, a3 = 0.f;
  const float4* W1v = reinterpret_cast<const float4*>(W1);  // row = 256 float4
#pragma unroll 4
  for (int s = 0; s < SEQ; ++s) {
    const float4 w = W1v[(size_t)toks[s] * (H1D / 4) + t];
    a0 += w.x; a1 += w.y; a2 += w.z; a3 += w.w;
  }
  const float4 bb = reinterpret_cast<const float4*>(b1)[t];
  a0 = fmaxf(a0 + bb.x, 0.f);
  a1 = fmaxf(a1 + bb.y, 0.f);
  a2 = fmaxf(a2 + bb.z, 0.f);
  a3 = fmaxf(a3 + bb.w, 0.f);

  v4bf o;
  o[0] = (bf16_t)a0; o[1] = (bf16_t)a1; o[2] = (bf16_t)a2; o[3] = (bf16_t)a3;
  *reinterpret_cast<v4bf*>(h1 + (size_t)b * H1D + t * 4) = o;
}

// ---------------------------------------------------------------------------
// Stage 1b: W2 [K=1024][N=512] f32  ->  W2t [N=512][K=1024] bf16.
// Transposed so each WMMA B-fragment lane reads contiguous 8-element K runs.
// ---------------------------------------------------------------------------
__global__ __launch_bounds__(256) void k1b_w2_cvt(
    const float* __restrict__ W2, bf16_t* __restrict__ w2t) {
  const int idx = blockIdx.x * 256 + threadIdx.x;   // 0 .. 512*1024-1
  const int n = idx >> 10;
  const int k = idx & 1023;
  w2t[idx] = (bf16_t)W2[(size_t)k * H2D + n];
}

// ---------------------------------------------------------------------------
// Stage 2: h2 = relu(h1 @ W2 + b2) via v_wmma_f32_16x16x32_bf16.
// One wave -> 32x32 output tile (4 WMMAs / K-step, each fragment reused 2x).
// 8 waves/block arranged 2(M) x 4(N) -> 64x128 block tile. grid = (4,16).
// ---------------------------------------------------------------------------
__global__ __launch_bounds__(256) void k2_fc2_wmma(
    const bf16_t* __restrict__ h1, const bf16_t* __restrict__ w2t,
    const float* __restrict__ b2, float* __restrict__ h2) {
  const int lane  = threadIdx.x & 31;
  const int wave  = threadIdx.x >> 5;       // 0..7
  const int waveN = wave & 3;
  const int waveM = wave >> 2;
  const int m0 = blockIdx.y * 64  + waveM * 32;
  const int n0 = blockIdx.x * 128 + waveN * 32;
  const int lh = lane & 15;                 // row/col within half
  const int hh = lane >> 4;                 // which K-half this lane carries

  v8f c00 = {}, c01 = {}, c10 = {}, c11 = {};

  // A-fragment (16x32 bf16): lane holds row m0+lh, K pieces [hh*8..+7] and [16+hh*8..+7]
  const bf16_t* aRow0 = h1 + (size_t)(m0 + lh)      * H1D + hh * 8;
  const bf16_t* aRow1 = h1 + (size_t)(m0 + 16 + lh) * H1D + hh * 8;
  // B-fragment (32x16 bf16): lane holds col n0+lh, K pieces [hh*8..+7] and [16+hh*8..+7]
  const bf16_t* bCol0 = w2t + (size_t)(n0 + lh)      * H1D + hh * 8;
  const bf16_t* bCol1 = w2t + (size_t)(n0 + 16 + lh) * H1D + hh * 8;

  for (int k = 0; k < H1D; k += 32) {
    union { v16bf v; v8bf h[2]; } fa0, fa1, fb0, fb1;
    fa0.h[0] = *(const v8bf*)(aRow0 + k);
    fa0.h[1] = *(const v8bf*)(aRow0 + k + 16);
    fa1.h[0] = *(const v8bf*)(aRow1 + k);
    fa1.h[1] = *(const v8bf*)(aRow1 + k + 16);
    fb0.h[0] = *(const v8bf*)(bCol0 + k);
    fb0.h[1] = *(const v8bf*)(bCol0 + k + 16);
    fb1.h[0] = *(const v8bf*)(bCol1 + k);
    fb1.h[1] = *(const v8bf*)(bCol1 + k + 16);

    c00 = __builtin_amdgcn_wmma_f32_16x16x32_bf16(false, fa0.v, false, fb0.v,
                                                  (short)0, c00, false, false);
    c01 = __builtin_amdgcn_wmma_f32_16x16x32_bf16(false, fa0.v, false, fb1.v,
                                                  (short)0, c01, false, false);
    c10 = __builtin_amdgcn_wmma_f32_16x16x32_bf16(false, fa1.v, false, fb0.v,
                                                  (short)0, c10, false, false);
    c11 = __builtin_amdgcn_wmma_f32_16x16x32_bf16(false, fa1.v, false, fb1.v,
                                                  (short)0, c11, false, false);
  }

  // Epilogue: bias + ReLU, store f32. C/D layout: vgpr r, lane L -> (m=r+8*hh, n=lh).
  const int n = n0 + lh;
  const float bias0 = b2[n];
  const float bias1 = b2[n + 16];
#pragma unroll
  for (int r = 0; r < 8; ++r) {
    const int m = m0 + r + hh * 8;
    h2[(size_t)m        * H2D + n]      = fmaxf(c00[r] + bias0, 0.f);
    h2[(size_t)m        * H2D + n + 16] = fmaxf(c01[r] + bias1, 0.f);
    h2[(size_t)(m + 16) * H2D + n]      = fmaxf(c10[r] + bias0, 0.f);
    h2[(size_t)(m + 16) * H2D + n + 16] = fmaxf(c11[r] + bias1, 0.f);
  }
}

// ---------------------------------------------------------------------------
// Stage 3: out[b,:] = h2[b,:] @ Wout + bout  (512x20, fp32). h2 row staged in
// LDS (broadcast reads), 20 threads each do one 512-MAC dot product.
// ---------------------------------------------------------------------------
__global__ __launch_bounds__(256) void k3_fcout(
    const float* __restrict__ h2, const float* __restrict__ Wout,
    const float* __restrict__ bout, float* __restrict__ out) {
  __shared__ float row[H2D];
  const int b = blockIdx.x;
  const int t = threadIdx.x;
  row[t]       = h2[(size_t)b * H2D + t];
  row[t + 256] = h2[(size_t)b * H2D + t + 256];
  __syncthreads();
  if (t < NCLS) {
    float acc = bout[t];
#pragma unroll 8
    for (int k = 0; k < H2D; ++k)
      acc = fmaf(row[k], Wout[k * NCLS + t], acc);
    out[b * NCLS + t] = acc;
  }
}

extern "C" void kernel_launch(void* const* d_in, const int* in_sizes, int n_in,
                              void* d_out, int out_size, void* d_ws, size_t ws_size,
                              hipStream_t stream) {
  (void)in_sizes; (void)n_in; (void)out_size; (void)ws_size;
  const int*   x    = (const int*)d_in[0];
  const float* W1   = (const float*)d_in[1];
  const float* b1   = (const float*)d_in[2];
  const float* W2   = (const float*)d_in[3];
  const float* b2   = (const float*)d_in[4];
  const float* Wout = (const float*)d_in[5];
  const float* bout = (const float*)d_in[6];
  float* out = (float*)d_out;

  char* ws = (char*)d_ws;
  bf16_t* h1  = (bf16_t*)ws;                                            // 2 MB
  bf16_t* w2t = (bf16_t*)(ws + (size_t)BATCH * H1D * sizeof(bf16_t));   // 1 MB
  float*  h2  = (float*)(ws + (size_t)BATCH * H1D * sizeof(bf16_t)
                            + (size_t)H2D   * H1D * sizeof(bf16_t));    // 2 MB

  k1_bow_fc1<<<dim3(BATCH), dim3(256), 0, stream>>>(x, W1, b1, h1);
  k1b_w2_cvt<<<dim3((H1D * H2D) / 256), dim3(256), 0, stream>>>(W2, w2t);
  k2_fc2_wmma<<<dim3(H2D / 128, BATCH / 64), dim3(256), 0, stream>>>(h1, w2t, b2, h2);
  k3_fcout<<<dim3(BATCH), dim3(256), 0, stream>>>(h2, Wout, bout, out);
}
